// CrossScaleAttention_31361851195657
// MI455X (gfx1250) — matrix operand order repaired and checked by reading, not compiled, not run on hardware
//
#include <hip/hip_runtime.h>
#include <hip/hip_bf16.h>

// ---------------------------------------------------------------------------
// CrossScaleAttention for MI455X (gfx1250, wave32, WMMA).
// B=8, C=256, N=4096, Cq=32.  ~96 GFLOP; fused flash-style attention keeps
// data movement to a few hundred MB (V fits in the 192MB L2), so bf16 WMMA
// compute and HBM/L2 traffic both land at tens of microseconds.
// Attention inner loop stages K+V tiles into LDS with the gfx1250 async
// copy path (global_load_async_to_lds_b128 + s_wait_asynccnt), double
// buffered, shared by all 4 waves; WMMA operands come from ds_load_b128.
// ---------------------------------------------------------------------------

typedef float v8f  __attribute__((ext_vector_type(8)));
typedef __bf16 v16bf __attribute__((ext_vector_type(16)));
typedef __bf16 bf16x8 __attribute__((ext_vector_type(8)));

#define NPIX 4096
#define CFULL 256
#define CQ 32
#define MTILES (NPIX / 32)

__device__ __forceinline__ bf16x8 load8f_bf(const float* p) {
  // 8 consecutive f32 -> 8 bf16 (two aligned float4 loads)
  float4 u = *(const float4*)p;
  float4 v = *(const float4*)(p + 4);
  bf16x8 r;
  r[0] = (__bf16)u.x; r[1] = (__bf16)u.y; r[2] = (__bf16)u.z; r[3] = (__bf16)u.w;
  r[4] = (__bf16)v.x; r[5] = (__bf16)v.y; r[6] = (__bf16)v.z; r[7] = (__bf16)v.w;
  return r;
}

__device__ __forceinline__ bf16x8 load8bf(const __bf16* p) {
  return *(const bf16x8*)p;   // 16B global_load_b128 / ds_load_b128
}

__device__ __forceinline__ v16bf cat8(bf16x8 lo, bf16x8 hi) {
  return __builtin_shufflevector(lo, hi, 0,1,2,3,4,5,6,7,8,9,10,11,12,13,14,15);
}

__device__ __forceinline__ float silu(float y) {
  return y / (1.0f + __expf(-y));
}

// gfx1250 async copy: global memory -> LDS, 16B per lane, tracked by ASYNCcnt.
// VDST = VGPR holding LDS byte address, VADDR = 64-bit global address.
__device__ __forceinline__ void async_ld_b128(unsigned int ldsoff, unsigned long long gaddr) {
  asm volatile("global_load_async_to_lds_b128 %0, %1, off"
               :: "v"(ldsoff), "v"(gaddr) : "memory");
}
__device__ __forceinline__ void wait_async_le9() {
  asm volatile("s_wait_asynccnt 0x9" ::: "memory");
}
__device__ __forceinline__ void wait_async_0() {
  asm volatile("s_wait_asynccnt 0x0" ::: "memory");
}

// ---------------------------------------------------------------------------
// Kernel 1: fused QKV projection.  y[o,n] = sum_c W[o,c] x[b,c,n], o in [0,320)
// (rows 0-31 = wq, 32-63 = wk, 64-319 = wv), then BN affine + SiLU.
// Writes Q,K as bf16 [B][N][32] (c contiguous -> WMMA operand friendly),
// V as bf16 [B][C][N] (n contiguous -> streamed into LDS by the attn kernel).
// One wave per 16x16 output tile; K-loop of 8 WMMAs (K=256).
// ---------------------------------------------------------------------------
__global__ void proj_kernel(const float* __restrict__ x,
                            const float* __restrict__ wq, const float* __restrict__ sq, const float* __restrict__ bq,
                            const float* __restrict__ wk, const float* __restrict__ sk, const float* __restrict__ bk,
                            const float* __restrict__ wv, const float* __restrict__ sv, const float* __restrict__ bv,
                            __bf16* __restrict__ Q, __bf16* __restrict__ K, __bf16* __restrict__ V)
{
  const int lane = threadIdx.x & 31;
  const int w    = threadIdx.x >> 5;
  int wid   = blockIdx.x * 4 + w;
  const int ntile = wid & 255;  wid >>= 8;   // 256 n-tiles
  const int otile = wid % 20;                // 20 o-tiles (320 rows)
  const int b     = wid / 20;
  const int half = lane >> 4, r = lane & 15;

  // A-operand row (per-lane) for W tile [16 o x 32 c]
  const int oa = otile * 16 + r;
  const float* wrow;
  if (oa < 32)      wrow = wq + (size_t)oa * CFULL;
  else if (oa < 64) wrow = wk + (size_t)(oa - 32) * CFULL;
  else              wrow = wv + (size_t)(oa - 64) * CFULL;

  const int n = ntile * 16 + r;              // B-operand column (per-lane)
  const float* xcol = x + (size_t)b * CFULL * NPIX + n;

  v8f acc = {0.f, 0.f, 0.f, 0.f, 0.f, 0.f, 0.f, 0.f};
#pragma unroll
  for (int kk = 0; kk < 8; ++kk) {
    const int c0 = kk * 32;
    // A: row oa, K chunks at c0+half*8 and c0+16+half*8
    v16bf A = cat8(load8f_bf(wrow + c0 + half * 8),
                   load8f_bf(wrow + c0 + 16 + half * 8));
    // B: column n, K values c0+half*8+0..7 and c0+16+half*8+0..7 (strided over x channels)
    bf16x8 b0, b1;
#pragma unroll
    for (int j = 0; j < 8; ++j) b0[j] = (__bf16)xcol[(size_t)(c0 + half * 8 + j) * NPIX];
#pragma unroll
    for (int j = 0; j < 8; ++j) b1[j] = (__bf16)xcol[(size_t)(c0 + 16 + half * 8 + j) * NPIX];
    v16bf Bm = cat8(b0, b1);
    acc = __builtin_amdgcn_wmma_f32_16x16x32_bf16(false, A, false, Bm, (short)0, acc, false, false);
  }

  // Epilogue: BN affine + SiLU, scatter to Q/K (transposed) or V.
  const int obase = otile * 16 + 8 * half;   // D-layout: VGPR g -> row obase+g, col = lane&15
#pragma unroll
  for (int g = 0; g < 8; ++g) {
    const int o = obase + g;
    float s, bb;
    if (o < 32)      { s = sq[o];      bb = bq[o];      }
    else if (o < 64) { s = sk[o - 32]; bb = bk[o - 32]; }
    else             { s = sv[o - 64]; bb = bv[o - 64]; }
    const float y = silu(acc[g] * s + bb);
    const __bf16 h = (__bf16)y;
    if (o < 32)      Q[((size_t)b * NPIX + n) * CQ + o]        = h;
    else if (o < 64) K[((size_t)b * NPIX + n) * CQ + (o - 32)] = h;
    else             V[((size_t)b * CFULL + (o - 64)) * NPIX + n] = h;
  }
}

// ---------------------------------------------------------------------------
// Kernel 2: flash attention.  Block = 4 waves = one 16-query tile; each wave
// owns 64 V channels (4 c-tiles); the block streams all 4096 keys in m-tiles
// of 32.  Per m-tile the block cooperatively async-copies the K tile (32x32
// bf16, 2KB) and V tile (256x32 bf16, 16KB) into a double-buffered LDS
// stage (9 global_load_async_to_lds_b128 per wave), then:
//   S[16n x 32m] = Q^T K      (2 WMMAs, operands via ds_load_b128)
//   online softmax over m     (shfl_xor row reductions in D-layout)
//   O[c,n] += V P^T           (4 WMMAs per wave)
// P / per-row rescale factors cross lane layouts via a per-wave LDS slice.
// ---------------------------------------------------------------------------
__global__ void attn_kernel(const __bf16* __restrict__ Q,
                            const __bf16* __restrict__ K,
                            const __bf16* __restrict__ V,
                            __bf16* __restrict__ AO)
{
  __shared__ __attribute__((aligned(16))) __bf16 Klds[2][32 * 32];     //  4 KB
  __shared__ __attribute__((aligned(16))) __bf16 Vlds[2][256 * 32];    // 32 KB
  __shared__ __attribute__((aligned(16))) __bf16 Plds[4][16 * 32];     //  4 KB
  __shared__ float Flds[4][16];

  const int t    = threadIdx.x;
  const int lane = t & 31;
  const int w    = t >> 5;
  const int qt   = blockIdx.x & 255;
  const int b    = blockIdx.x >> 8;
  const int half = lane >> 4, r = lane & 15;
  const int cbase = w * 64;

  const __bf16* Kb = K + (size_t)b * NPIX * CQ;
  const __bf16* Vb = V + (size_t)b * CFULL * NPIX;

  // Q A-operand tile [16 n x 32 c], loaded once (rows = queries, K = channels)
  const int qn = qt * 16 + r;
  const __bf16* qrow = Q + ((size_t)b * NPIX + qn) * CQ;
  const v16bf QA = cat8(load8bf(qrow + half * 16), load8bf(qrow + half * 16 + 8));

  v8f O[4];
#pragma unroll
  for (int ct = 0; ct < 4; ++ct) O[ct] = (v8f){0.f,0.f,0.f,0.f,0.f,0.f,0.f,0.f};
  float rm[8], ls[8];
#pragma unroll
  for (int g = 0; g < 8; ++g) { rm[g] = -3.0e38f; ls[g] = 0.0f; }

  // Cooperative async tile copy: 9 x b128 per thread-slot, 18KB per m-tile.
  const int krow = t >> 2, kch = t & 3;        // K tile: 128 chunks of 16B
  auto issue_tile = [&](int buf, int m0) {
    async_ld_b128((unsigned int)(size_t)&Klds[buf][krow * 32 + kch * 8],
                  (unsigned long long)(size_t)(Kb + (size_t)(m0 + krow) * CQ + kch * 8));
#pragma unroll
    for (int i = 0; i < 8; ++i) {              // V tile: 1024 chunks of 16B
      const int e   = i * 128 + t;
      const int row = e >> 2, ch = e & 3;
      async_ld_b128((unsigned int)(size_t)&Vlds[buf][row * 32 + ch * 8],
                    (unsigned long long)(size_t)(Vb + (size_t)row * NPIX + m0 + ch * 8));
    }
  };

  issue_tile(0, 0);

  for (int mt = 0; mt < MTILES; ++mt) {
    const int cur = mt & 1;
    if (mt + 1 < MTILES) {
      issue_tile(cur ^ 1, (mt + 1) * 32);      // prefetch next tile into other buffer
      wait_async_le9();                        // current tile's 9 loads complete
    } else {
      wait_async_0();
    }
    __syncthreads();                           // tile visible block-wide

    // K B-operand tiles from LDS: column m_local, K = c chunk half*16
    const __bf16* k0p = &Klds[cur][r * 32 + half * 16];
    const __bf16* k1p = &Klds[cur][(16 + r) * 32 + half * 16];
    const v16bf B0 = cat8(load8bf(k0p), load8bf(k0p + 8));
    const v16bf B1 = cat8(load8bf(k1p), load8bf(k1p + 8));

    v8f S0 = {0.f,0.f,0.f,0.f,0.f,0.f,0.f,0.f};
    v8f S1 = {0.f,0.f,0.f,0.f,0.f,0.f,0.f,0.f};
    S0 = __builtin_amdgcn_wmma_f32_16x16x32_bf16(false, QA, false, B0, (short)0, S0, false, false);
    S1 = __builtin_amdgcn_wmma_f32_16x16x32_bf16(false, QA, false, B1, (short)0, S1, false, false);

    // Online softmax per query row (row n = g + 8*half in D-layout; the 32
    // m-values of a row live across the 16 lanes of one half x two tiles).
#pragma unroll
    for (int g = 0; g < 8; ++g) {
      float tmx = fmaxf(S0[g], S1[g]);
      tmx = fmaxf(tmx, __shfl_xor(tmx, 1));
      tmx = fmaxf(tmx, __shfl_xor(tmx, 2));
      tmx = fmaxf(tmx, __shfl_xor(tmx, 4));
      tmx = fmaxf(tmx, __shfl_xor(tmx, 8));
      const float nm   = fmaxf(rm[g], tmx);
      const float corr = __expf(rm[g] - nm);
      rm[g] = nm;
      const float p0 = __expf(S0[g] - nm);
      const float p1 = __expf(S1[g] - nm);
      float rs = p0 + p1;
      rs += __shfl_xor(rs, 1);
      rs += __shfl_xor(rs, 2);
      rs += __shfl_xor(rs, 4);
      rs += __shfl_xor(rs, 8);
      ls[g] = ls[g] * corr + rs;
      if (r == g) Flds[w][half * 8 + g] = corr;          // per-row rescale factor
      const int nrow = half * 8 + g;
      Plds[w][nrow * 32 + r]      = (__bf16)p0;          // P in [n][m] layout
      Plds[w][nrow * 32 + 16 + r] = (__bf16)p1;
    }

    // Rescale accumulators (O cols are queries: n_local = lane&15)
    const float f = Flds[w][r];
#pragma unroll
    for (int ct = 0; ct < 4; ++ct) O[ct] *= f;

    // P^T as B-operand: column n = lane&15, K = 16 m-values from half*16
    const __bf16* pl = &Plds[w][r * 32 + half * 16];
    const v16bf PB = cat8(load8bf(pl), load8bf(pl + 8));

    // O[16c x 16n] += V[16c x 32m] * P^T[32m x 16n], 4 channel tiles per wave
#pragma unroll
    for (int ct = 0; ct < 4; ++ct) {
      const __bf16* vp = &Vlds[cur][(cbase + ct * 16 + r) * 32 + half * 16];
      const v16bf VA = cat8(load8bf(vp), load8bf(vp + 8));
      O[ct] = __builtin_amdgcn_wmma_f32_16x16x32_bf16(false, VA, false, PB, (short)0, O[ct], false, false);
    }

    __syncthreads();                           // all waves done with buf[cur]
  }

  // Normalize by the softmax denominator (per query row -> per-lane via LDS)
#pragma unroll
  for (int g = 0; g < 8; ++g) {
    const float inv = 1.0f / ls[g];
    if (r == g) Flds[w][half * 8 + g] = inv;
  }
  const float f = Flds[w][r];
#pragma unroll
  for (int ct = 0; ct < 4; ++ct) O[ct] *= f;

  // Store attn output bf16 as [B][N][C] (c contiguous for the next GEMM's B)
  const int n = qt * 16 + r;
  __bf16* ao = AO + ((size_t)b * NPIX + n) * CFULL;
#pragma unroll
  for (int ct = 0; ct < 4; ++ct)
#pragma unroll
    for (int g = 0; g < 8; ++g)
      ao[cbase + ct * 16 + 8 * half + g] = (__bf16)O[ct][g];
}

// ---------------------------------------------------------------------------
// Kernel 3: output projection y = SiLU(BN(wo @ AO)); out = gamma*y + x.
// One wave per 16x16 tile over [256 o x 4096 n], K-loop of 8 WMMAs (K=256).
// ---------------------------------------------------------------------------
__global__ void out_kernel(const float* __restrict__ wo, const float* __restrict__ so,
                           const float* __restrict__ bo, const float* __restrict__ gamma,
                           const float* __restrict__ x,  const __bf16* __restrict__ AO,
                           float* __restrict__ out)
{
  const int lane = threadIdx.x & 31;
  const int w    = threadIdx.x >> 5;
  int wid   = blockIdx.x * 4 + w;
  const int ntile = wid & 255;  wid >>= 8;
  const int otile = wid & 15;
  const int b     = wid >> 4;
  const int half = lane >> 4, r = lane & 15;

  const int n = ntile * 16 + r;
  const float* wrow = wo + (size_t)(otile * 16 + r) * CFULL;
  const __bf16* arow = AO + ((size_t)b * NPIX + n) * CFULL;

  v8f acc = {0.f,0.f,0.f,0.f,0.f,0.f,0.f,0.f};
#pragma unroll
  for (int kk = 0; kk < 8; ++kk) {
    const int c0 = kk * 32;
    v16bf A = cat8(load8f_bf(wrow + c0 + half * 8),
                   load8f_bf(wrow + c0 + 16 + half * 8));
    const __bf16* bp = arow + c0 + half * 8;
    v16bf Bm = cat8(load8bf(bp), load8bf(bp + 16));
    acc = __builtin_amdgcn_wmma_f32_16x16x32_bf16(false, A, false, Bm, (short)0, acc, false, false);
  }

  const float gm = gamma[0];
  const int obase = otile * 16 + 8 * half;
#pragma unroll
  for (int g = 0; g < 8; ++g) {
    const int o = obase + g;
    const float y = silu(acc[g] * so[o] + bo[o]);
    const size_t idx = ((size_t)b * CFULL + o) * NPIX + n;
    out[idx] = gm * y + x[idx];
  }
}

// ---------------------------------------------------------------------------
// Launch: 3 kernels on `stream`.  Workspace layout (bf16):
//   Q  [8][4096][32]   @ 0        (2 MiB)
//   K  [8][4096][32]   @ 2 MiB    (2 MiB)
//   V  [8][256][4096]  @ 4 MiB    (16 MiB)
//   AO [8][4096][256]  @ 20 MiB   (16 MiB)   total 36 MiB
// ---------------------------------------------------------------------------
extern "C" void kernel_launch(void* const* d_in, const int* in_sizes, int n_in,
                              void* d_out, int out_size, void* d_ws, size_t ws_size,
                              hipStream_t stream) {
  const float* x  = (const float*)d_in[0];
  const float* wq = (const float*)d_in[1];
  const float* sq = (const float*)d_in[2];
  const float* bq = (const float*)d_in[3];
  const float* wk = (const float*)d_in[4];
  const float* sk = (const float*)d_in[5];
  const float* bk = (const float*)d_in[6];
  const float* wv = (const float*)d_in[7];
  const float* sv = (const float*)d_in[8];
  const float* bv = (const float*)d_in[9];
  const float* wo = (const float*)d_in[10];
  const float* so = (const float*)d_in[11];
  const float* bo = (const float*)d_in[12];
  const float* gamma = (const float*)d_in[13];

  char* ws = (char*)d_ws;
  __bf16* Q  = (__bf16*)(ws);
  __bf16* K  = (__bf16*)(ws + (size_t)(2)  * 1024 * 1024);
  __bf16* V  = (__bf16*)(ws + (size_t)(4)  * 1024 * 1024);
  __bf16* AO = (__bf16*)(ws + (size_t)(20) * 1024 * 1024);

  // 8 batches * 20 o-tiles * 256 n-tiles = 40960 tiles, 4 waves/block
  proj_kernel<<<10240, 128, 0, stream>>>(x, wq, sq, bq, wk, sk, bk, wv, sv, bv, Q, K, V);
  // 8 batches * 256 query tiles
  attn_kernel<<<2048, 128, 0, stream>>>(Q, K, V, AO);
  // 8 batches * 16 o-tiles * 256 n-tiles = 32768 tiles, 4 waves/block
  out_kernel<<<8192, 128, 0, stream>>>(wo, so, bo, gamma, x, AO, (float*)d_out);
}
